// PatternLayer_83451214561454
// MI455X (gfx1250) — matrix operand order repaired and checked by reading, not compiled, not run on hardware
//
#include <hip/hip_runtime.h>

typedef __attribute__((ext_vector_type(16))) __bf16 v16bf;
typedef __attribute__((ext_vector_type(8)))  __bf16 v8bf;
typedef __attribute__((ext_vector_type(8)))  float  v8f;

#define D_DIM    1024
#define TILE_M   128
#define TILE_N   128
#define TILE_K   32
#define A_STRIDE 40    // bf16 elems per M-row (32 + 8 pad) -> 80 bytes
#define B_STRIDE 136   // bf16 elems per K-row (128 + 8 pad) -> 272 bytes

static __device__ __forceinline__ unsigned int bfbits(__bf16 v) {
  return (unsigned int)__builtin_bit_cast(unsigned short, v);
}

#define WMMA_BF16(a, b, c) \
  __builtin_amdgcn_wmma_f32_16x16x32_bf16(false, (a), false, (b), (short)0, (c), false, false)

__global__ __launch_bounds__(256)
void rbf_wmma_kernel(const float* __restrict__ X, const float* __restrict__ T,
                     float* __restrict__ Out, int Ndim)
{
  __shared__ __bf16 Ah[TILE_M * A_STRIDE];
  __shared__ __bf16 Al[TILE_M * A_STRIDE];
  __shared__ __bf16 Bh[TILE_K * B_STRIDE];
  __shared__ __bf16 Bl[TILE_K * B_STRIDE];
  __shared__ float  Xs[TILE_M];
  __shared__ float  Ts[TILE_N];

  const int tid  = threadIdx.x;
  const int lane = tid & 31;           // wave32
  const int wave = tid >> 5;           // 0..7
  const int wm   = wave >> 2;          // 0..1 : wave M position (64 rows each)
  const int wn   = wave & 3;           // 0..3 : wave N position (32 cols each)

  const int bm0 = blockIdx.y * TILE_M;
  const int bn0 = blockIdx.x * TILE_N;

  const int row = tid >> 1;            // 0..127: row handled for staging
  const int kh  = (tid & 1) * 16;      // K half: 0 or 16

  const float* xptr = X + (size_t)(bm0 + row) * D_DIM + kh;
  const float* tptr = T + (size_t)(bn0 + row) * D_DIM + kh;

  float xsq = 0.f, tsq = 0.f;

  v8f acc[4][2];
  #pragma unroll
  for (int mi = 0; mi < 4; ++mi)
    #pragma unroll
    for (int ni = 0; ni < 2; ++ni)
      #pragma unroll
      for (int e = 0; e < 8; ++e)
        acc[mi][ni][e] = 0.0f;

  for (int k0 = 0; k0 < D_DIM; k0 += TILE_K) {
    // -------- global stage: 16 f32 of one x-row and one t-row per thread
    float xv[16], tv[16];
    #pragma unroll
    for (int i = 0; i < 4; ++i) {
      const float4 a = *(const float4*)(xptr + k0 + 4*i);
      const float4 b = *(const float4*)(tptr + k0 + 4*i);
      xv[4*i+0]=a.x; xv[4*i+1]=a.y; xv[4*i+2]=a.z; xv[4*i+3]=a.w;
      tv[4*i+0]=b.x; tv[4*i+1]=b.y; tv[4*i+2]=b.z; tv[4*i+3]=b.w;
    }

    __syncthreads();   // protect previous iteration's LDS fragment reads

    // -------- split f32 -> bf16 hi/lo, accumulate squared norms, fill LDS
    {
      unsigned int pah[8], pal[8];
      #pragma unroll
      for (int j = 0; j < 8; ++j) {
        const float x0 = xv[2*j], x1 = xv[2*j+1];
        xsq += x0*x0 + x1*x1;
        const __bf16 h0 = (__bf16)x0; const __bf16 l0 = (__bf16)(x0 - (float)h0);
        const __bf16 h1 = (__bf16)x1; const __bf16 l1 = (__bf16)(x1 - (float)h1);
        pah[j] = bfbits(h0) | (bfbits(h1) << 16);
        pal[j] = bfbits(l0) | (bfbits(l1) << 16);
      }
      uint4* pa = (uint4*)((char*)Ah + row*(A_STRIDE*2) + kh*2);
      pa[0] = make_uint4(pah[0], pah[1], pah[2], pah[3]);
      pa[1] = make_uint4(pah[4], pah[5], pah[6], pah[7]);
      uint4* pl = (uint4*)((char*)Al + row*(A_STRIDE*2) + kh*2);
      pl[0] = make_uint4(pal[0], pal[1], pal[2], pal[3]);
      pl[1] = make_uint4(pal[4], pal[5], pal[6], pal[7]);

      // B stored transposed into WMMA-B order: B[k][n], n contiguous
      #pragma unroll
      for (int j = 0; j < 16; ++j) {
        const float t0 = tv[j];
        tsq += t0*t0;
        const __bf16 h = (__bf16)t0; const __bf16 l = (__bf16)(t0 - (float)h);
        Bh[(kh + j)*B_STRIDE + row] = h;
        Bl[(kh + j)*B_STRIDE + row] = l;
      }
    }

    __syncthreads();   // LDS tiles ready

    // -------- load fragments per ISA layout
    v16bf afh[4], afl[4], bfrh[2], bfrl[2];
    const int ar     = lane & 15;            // M within 16-tile
    const int achunk = (lane >> 4) * 16;     // byte offset: K0..7 vs K8..15
    #pragma unroll
    for (int mi = 0; mi < 4; ++mi) {
      const int r = wm*64 + mi*16 + ar;
      const char* ph = (const char*)Ah + r*(A_STRIDE*2) + achunk;
      const char* pl = (const char*)Al + r*(A_STRIDE*2) + achunk;
      const v8bf h0 = *(const v8bf*)ph;
      const v8bf h1 = *(const v8bf*)(ph + 32);
      const v8bf l0 = *(const v8bf*)pl;
      const v8bf l1 = *(const v8bf*)(pl + 32);
      afh[mi] = __builtin_shufflevector(h0, h1, 0,1,2,3,4,5,6,7,8,9,10,11,12,13,14,15);
      afl[mi] = __builtin_shufflevector(l0, l1, 0,1,2,3,4,5,6,7,8,9,10,11,12,13,14,15);
    }
    #pragma unroll
    for (int ni = 0; ni < 2; ++ni) {
      const int bn = wn*32 + ni*16 + (lane & 15);   // N column for this lane
      const int kb = (lane >> 4) * 8;
      #pragma unroll
      for (int s = 0; s < 16; ++s) {
        const int k = (s < 8) ? (kb + s) : (16 + kb + (s - 8));
        bfrh[ni][s] = Bh[k*B_STRIDE + bn];
        bfrl[ni][s] = Bl[k*B_STRIDE + bn];
      }
    }

    // -------- 3-term bf16 split GEMM: hh + hl + lh, accumulated in f32
    #pragma unroll
    for (int mi = 0; mi < 4; ++mi)
      #pragma unroll
      for (int ni = 0; ni < 2; ++ni) {
        acc[mi][ni] = WMMA_BF16(afh[mi], bfrh[ni], acc[mi][ni]);
        acc[mi][ni] = WMMA_BF16(afh[mi], bfrl[ni], acc[mi][ni]);
        acc[mi][ni] = WMMA_BF16(afl[mi], bfrh[ni], acc[mi][ni]);
      }
  }

  // -------- finalize row norms (pair reduce: threads 2r, 2r+1 share a row)
  const float xall = xsq + __shfl_xor(xsq, 1, 32);
  const float tall = tsq + __shfl_xor(tsq, 1, 32);
  if ((tid & 1) == 0) { Xs[row] = xall; Ts[row] = tall; }
  __syncthreads();

  // -------- epilogue: dist = xs + ts - 2c, clamp, exp(-dist/2)
  #pragma unroll
  for (int mi = 0; mi < 4; ++mi) {
    #pragma unroll
    for (int ni = 0; ni < 2; ++ni) {
      const int ln  = wn*32 + ni*16 + (lane & 15);
      const float tn = Ts[ln];
      const int lmb = wm*64 + mi*16 + (lane >> 4) * 8;
      #pragma unroll
      for (int v = 0; v < 8; ++v) {
        const int lm = lmb + v;
        float dist = Xs[lm] + tn - 2.0f * acc[mi][ni][v];
        dist = fmaxf(dist, 0.0f);
        Out[(size_t)(bm0 + lm) * (size_t)Ndim + (size_t)(bn0 + ln)] = __expf(-0.5f * dist);
      }
    }
  }
}

extern "C" void kernel_launch(void* const* d_in, const int* in_sizes, int n_in,
                              void* d_out, int out_size, void* d_ws, size_t ws_size,
                              hipStream_t stream) {
  const float* X = (const float*)d_in[0];   // inputs  [B, 1024] f32
  const float* T = (const float*)d_in[1];   // train   [N, 1024] f32
  float* Out = (float*)d_out;               // [B, N] f32
  const int Bdim = in_sizes[0] / D_DIM;     // 4096
  const int Ndim = in_sizes[1] / D_DIM;     // 16384
  dim3 grid(Ndim / TILE_N, Bdim / TILE_M);  // (128, 32)
  rbf_wmma_kernel<<<grid, dim3(256, 1, 1), 0, stream>>>(X, T, Out, Ndim);
}
